// Decoder_66898410602723
// MI455X (gfx1250) — matrix operand order repaired and checked by reading, not compiled, not run on hardware
//
#include <hip/hip_runtime.h>
#include <hip/hip_bf16.h>

// ---------------------------------------------------------------------------
// CDNA5 (gfx1250) persistent single-WGP attention decoder.
// wave32, WMMA f32_16x16x32_f16 for GRU / location-conv / vocab projection.
// All recurrent state lives in LDS (~181 KB of the WGP's 320 KB).
// ---------------------------------------------------------------------------

typedef __attribute__((ext_vector_type(16))) _Float16 v16h;
typedef __attribute__((ext_vector_type(8)))  _Float16 v8h;
typedef __attribute__((ext_vector_type(8)))  float    v8f;

#define SHUF16(a,b) __builtin_shufflevector((a),(b),0,1,2,3,4,5,6,7,8,9,10,11,12,13,14,15)

constexpr int B_  = 8;
constexpr int T_  = 512;
constexpr int H_  = 256;
constexpr int L_  = 65;
constexpr int KS_ = 15;
constexpr int G3H = 3 * H_;      // 768
constexpr int NVP = 1008;        // V-1=999 padded to 63*16
constexpr int AXW = 560;         // padded attention row: 8 left pad + 512 + 40 right pad

// f16 weight staging in workspace
constexpr int WIH16_ELEMS = G3H * H_;     // 196608
constexpr int WHH16_ELEMS = G3H * H_;     // 196608
constexpr int FC16_ELEMS  = NVP * H_;     // 258048

// LDS carve (bytes): float region then f16 region, all 16B aligned.
constexpr int SMEM_FLOATS = 2048 + 2048 + 6144 + 6144 + 4096 + 4480 + 8064
                          + 256 + 256 + 768 + 768 + 1008 + 16;   // 36096
constexpr int SMEM_HALVES = 4096 + 4096 + 4096 + 8192;           // 20480
constexpr int SMEM_BYTES  = SMEM_FLOATS * 4 + SMEM_HALVES * 2;   // 185344

__global__ void cvt_weights_kernel(const float* __restrict__ w_ih,
                                   const float* __restrict__ w_hh,
                                   const float* __restrict__ fc_w,
                                   _Float16* __restrict__ w_ih16,
                                   _Float16* __restrict__ w_hh16,
                                   _Float16* __restrict__ fc16) {
    int i = blockIdx.x * blockDim.x + threadIdx.x;
    if (i < WIH16_ELEMS) {
        w_ih16[i] = (_Float16)w_ih[i];
        w_hh16[i] = (_Float16)w_hh[i];
    }
    if (i < FC16_ELEMS) {
        int row = i >> 8;                 // /H_
        fc16[i] = (row < 999) ? (_Float16)fc_w[i] : (_Float16)0.f;
    }
}

__global__ void __launch_bounds__(1024)
decoder_persistent_kernel(const int*   __restrict__ target,
                          const float* __restrict__ enc_out,
                          const float* __restrict__ enc_hid,
                          const float* __restrict__ embedding,
                          const float* __restrict__ b_ih,
                          const float* __restrict__ b_hh,
                          const float* __restrict__ conv_w,
                          const float* __restrict__ conv_b,
                          const float* __restrict__ attn_w,
                          const float* __restrict__ attn_b,
                          const float* __restrict__ fc_b,
                          const _Float16* __restrict__ w_ih16,
                          const _Float16* __restrict__ w_hh16,
                          const _Float16* __restrict__ fc16,
                          float* __restrict__ out) {
    extern __shared__ char smem[];
    float* hbuf   = (float*)smem;            // 8x256   current hidden state
    float* sx     = hbuf   + 2048;           // 8x256   attention context
    float* gi     = sx     + 2048;           // 8x768   input gates
    float* gh     = gi     + 6144;           // 8x768   hidden gates
    float* score  = gh     + 6144;           // 8x512   attention logits
    float* axpad  = score  + 4096;           // 8x560   padded attention weights
    float* logits = axpad  + 4480;           // 8x1008  vocab logits
    float* attnw  = logits + 8064;           // 256
    float* convb  = attnw  + 256;            // 256
    float* bihs   = convb  + 256;            // 768
    float* bhhs   = bihs   + 768;            // 768
    float* fcb    = bhhs   + 768;            // 1008
    float* nll    = fcb    + 1008;           // 16 (scalar + pad)
    _Float16* h16    = (_Float16*)(nll + 16);   // 16x256 (rows 8..15 zero)
    _Float16* emb16  = h16   + 4096;            // 16x256
    _Float16* hs16   = emb16 + 4096;            // 16x256 (h + sx)
    _Float16* conv16 = hs16  + 4096;            // 256x32 conv taps, K padded 15->32

    const int tid  = threadIdx.x;
    const int lane = tid & 31;
    const int wave = tid >> 5;
    const int tl   = lane & 15;   // WMMA row/col-in-tile index
    const int hi16 = lane >> 4;   // WMMA lane-half select
    const float logT = 6.2383246250395075f;   // log(512)

    // ---------------- one-time init ----------------
    for (int i = tid; i < 4096; i += 1024) {
        h16[i] = (_Float16)0.f; emb16[i] = (_Float16)0.f; hs16[i] = (_Float16)0.f;
    }
    for (int i = tid; i < 2048; i += 1024) {
        float v = enc_hid[i];
        hbuf[i] = v;
        h16[i]  = (_Float16)v;    // rows 0..7 contiguous
    }
    for (int i = tid; i < 256 * 32; i += 1024) {
        int hh = i >> 5, k = i & 31;
        conv16[i] = (k < KS_) ? (_Float16)conv_w[hh * KS_ + k] : (_Float16)0.f;
    }
    for (int i = tid; i < 8 * AXW; i += 1024) axpad[i] = 0.f;
    if (tid < 256) { attnw[tid] = attn_w[tid]; convb[tid] = conv_b[tid]; }
    if (tid < 768) { bihs[tid] = b_ih[tid];   bhhs[tid] = b_hh[tid]; }
    if (tid < NVP) fcb[tid] = (tid < 999) ? fc_b[tid] : 0.f;
    if (tid == 0)  nll[0] = 0.f;
    __syncthreads();
    const float attn_b0 = attn_b[0];

    // ---------------- 64 sequential decoder steps ----------------
    for (int s = 0; s < L_ - 1; ++s) {
        // Phase 1: embedding lookup (+ previous context) -> f16 A operand
        for (int i = tid; i < 2048; i += 1024) {
            int b = i >> 8, j = i & 255;
            int tg = target[b * L_ + s];
            float v = embedding[tg * H_ + j] + (s > 0 ? sx[i] : 0.f);
            emb16[i] = (_Float16)v;
        }
        __syncthreads();

        // Phase 2: GRU gate matmuls, 96 WMMA tiles (48 gi + 48 gh), 3/wave
        for (int tt = wave; tt < 96; tt += 32) {
            const int which = tt / 48;         // 0: gi(emb,w_ih)  1: gh(h,w_hh)
            const int o0 = (tt % 48) * 16;
            const _Float16* wq = which ? w_hh16 : w_ih16;
            const _Float16* aq = which ? h16 : emb16;
            v8f acc = {};
#pragma unroll
            for (int k0 = 0; k0 < H_; k0 += 32) {
                const v8h* pa0 = (const v8h*)(aq + tl * H_ + k0 + 8 * hi16);
                const v8h* pa1 = (const v8h*)(aq + tl * H_ + k0 + 16 + 8 * hi16);
                v16h am = SHUF16(pa0[0], pa1[0]);
                const v8h* pb0 = (const v8h*)(wq + (o0 + tl) * H_ + k0 + 8 * hi16);
                const v8h* pb1 = (const v8h*)(wq + (o0 + tl) * H_ + k0 + 16 + 8 * hi16);
                v16h bm = SHUF16(pb0[0], pb1[0]);
                acc = __builtin_amdgcn_wmma_f32_16x16x32_f16(
                          false, am, false, bm, (short)0, acc, false, false);
            }
            // Rows b = r + 8*hi16: only hi16==0 lanes hold valid batch rows.
            if (hi16 == 0) {
                float* gout = which ? gh : gi;
                const float bv = (which ? bhhs : bihs)[o0 + tl];
#pragma unroll
                for (int r = 0; r < 8; ++r)
                    gout[r * G3H + o0 + tl] = acc[r] + bv;
            }
        }
        __syncthreads();

        // Phase 3: GRU gating -> new hidden state (f32 + f16 copies)
        for (int i = tid; i < 2048; i += 1024) {
            int b = i >> 8, j = i & 255;
            float ir = gi[b * G3H + j],       hr = gh[b * G3H + j];
            float iz = gi[b * G3H + 256 + j], hz = gh[b * G3H + 256 + j];
            float in_ = gi[b * G3H + 512 + j], hn = gh[b * G3H + 512 + j];
            float r = 1.f / (1.f + __expf(-(ir + hr)));
            float z = 1.f / (1.f + __expf(-(iz + hz)));
            float n = tanhf(in_ + r * hn);
            float hv = (1.f - z) * n + z * hbuf[i];
            hbuf[i] = hv;
            h16[i]  = (_Float16)hv;
        }
        __syncthreads();

        // Phase 4: attention scores
        if (s == 0) {
            // no conv term: wave per (b,t), lane strides h
            for (int p = wave; p < B_ * T_; p += 32) {
                int b = p >> 9, t = p & 511;
                const float* er = enc_out + (size_t)(b * T_ + t) * H_;
                const float* hr = hbuf + b * H_;
                float acc = 0.f;
#pragma unroll
                for (int k = 0; k < 8; ++k) {
                    int hh = lane + 32 * k;
                    acc += fmaxf(er[hh] + hr[hh], 0.f) * attnw[hh];
                }
#pragma unroll
                for (int d = 16; d >= 1; d >>= 1) acc += __shfl_xor(acc, d, 32);
                if (lane == 0) score[b * T_ + t] = acc + attn_b0;
            }
        } else {
            // location conv as WMMA: C[h,t] = sum_K conv16[h,K] * ax[t+K-7]
            for (int p = wave; p < 256; p += 32) {
                int b = p >> 5, ttile = p & 31;
                int t = ttile * 16 + tl;
                // B operand: per-lane ax taps ar[0..7] and ar[16..23] (dword-aligned f32 gather)
                const float* ar = axpad + b * AXW + 1 + t + 8 * hi16;  // 8+(t-7)+8*hi16
                v16h bm = {};
#pragma unroll
                for (int j = 0; j < 16; ++j) bm[j] = (_Float16)ar[j < 8 ? j : j + 8];
                float sacc = 0.f;
                const float* er = enc_out + (size_t)(b * T_ + t) * H_;
                const float* hr = hbuf + b * H_;
                for (int ht = 0; ht < 16; ++ht) {
                    const v8h* pa0 = (const v8h*)(conv16 + (ht * 16 + tl) * 32 + 8 * hi16);
                    const v8h* pa1 = (const v8h*)(conv16 + (ht * 16 + tl) * 32 + 16 + 8 * hi16);
                    v16h am = SHUF16(pa0[0], pa1[0]);
                    v8f c = {};
                    c = __builtin_amdgcn_wmma_f32_16x16x32_f16(
                            false, am, false, bm, (short)0, c, false, false);
#pragma unroll
                    for (int r = 0; r < 8; ++r) {
                        int hh = ht * 16 + r + 8 * hi16;
                        float v = c[r] + er[hh] + hr[hh] + convb[hh];
                        sacc += fmaxf(v, 0.f) * attnw[hh];
                    }
                }
                sacc += __shfl_xor(sacc, 16, 32);   // join the two h-halves
                if (lane < 16) score[b * T_ + ttile * 16 + lane] = sacc + attn_b0;
            }
        }
        __syncthreads();

        // Phase 5: temperature softmax over T (one wave per batch row)
        if (wave < 8) {
            int b = wave;
            float ev[16];
            float m = -3.4e38f;
#pragma unroll
            for (int k = 0; k < 16; ++k) {
                float e = score[b * T_ + lane + 32 * k] * logT;
                ev[k] = e; m = fmaxf(m, e);
            }
#pragma unroll
            for (int d = 16; d >= 1; d >>= 1) m = fmaxf(m, __shfl_xor(m, d, 32));
            float sum = 0.f;
#pragma unroll
            for (int k = 0; k < 16; ++k) { ev[k] = __expf(ev[k] - m); sum += ev[k]; }
#pragma unroll
            for (int d = 16; d >= 1; d >>= 1) sum += __shfl_xor(sum, d, 32);
            float inv = 1.f / sum;
#pragma unroll
            for (int k = 0; k < 16; ++k)
                axpad[b * AXW + 8 + lane + 32 * k] = ev[k] * inv;
        }
        __syncthreads();

        // Phase 6: context sx = ax . enc_out ; fuse h+sx -> f16 FC A operand.
        // float4 over h, thread-pairs split the t-range, wave32 shfl combine.
        {
            const int task = tid >> 1;        // 512 tasks: (b, h-quad)
            const int half = tid & 1;         // t-range half
            const int b    = task >> 6;
            const int h0   = (task & 63) * 4;
            const float4* er4 = (const float4*)(enc_out + (size_t)b * T_ * H_ + h0);
            const float*  ax  = axpad + b * AXW + 8 + half * 256;
            const float4* e   = er4 + (size_t)half * 256 * (H_ / 4);
            float a0 = 0.f, a1 = 0.f, a2 = 0.f, a3 = 0.f;
#pragma unroll 4
            for (int t = 0; t < 256; ++t) {
                float w = ax[t];
                float4 ev = e[(size_t)t * (H_ / 4)];
                a0 += w * ev.x; a1 += w * ev.y; a2 += w * ev.z; a3 += w * ev.w;
            }
            a0 += __shfl_xor(a0, 1, 32);
            a1 += __shfl_xor(a1, 1, 32);
            a2 += __shfl_xor(a2, 1, 32);
            a3 += __shfl_xor(a3, 1, 32);
            if (half == 0) {
                const int base = b * H_ + h0;
                sx[base + 0] = a0; sx[base + 1] = a1;
                sx[base + 2] = a2; sx[base + 3] = a3;
                hs16[base + 0] = (_Float16)(hbuf[base + 0] + a0);
                hs16[base + 1] = (_Float16)(hbuf[base + 1] + a1);
                hs16[base + 2] = (_Float16)(hbuf[base + 2] + a2);
                hs16[base + 3] = (_Float16)(hbuf[base + 3] + a3);
            }
        }
        __syncthreads();

        // Phase 7: vocab projection, 63 WMMA tiles
        for (int nt = wave; nt < 63; nt += 32) {
            int o0 = nt * 16;
            v8f acc = {};
#pragma unroll
            for (int k0 = 0; k0 < H_; k0 += 32) {
                const v8h* pa0 = (const v8h*)(hs16 + tl * H_ + k0 + 8 * hi16);
                const v8h* pa1 = (const v8h*)(hs16 + tl * H_ + k0 + 16 + 8 * hi16);
                v16h am = SHUF16(pa0[0], pa1[0]);
                const v8h* pb0 = (const v8h*)(fc16 + (o0 + tl) * H_ + k0 + 8 * hi16);
                const v8h* pb1 = (const v8h*)(fc16 + (o0 + tl) * H_ + k0 + 16 + 8 * hi16);
                v16h bm = SHUF16(pb0[0], pb1[0]);
                acc = __builtin_amdgcn_wmma_f32_16x16x32_f16(
                          false, am, false, bm, (short)0, acc, false, false);
            }
            if (hi16 == 0) {
                const float bv = fcb[o0 + tl];
#pragma unroll
                for (int r = 0; r < 8; ++r)
                    logits[r * NVP + o0 + tl] = acc[r] + bv;
            }
        }
        __syncthreads();

        // Phase 8: log-softmax NLL accumulate (one wave per batch row)
        if (wave < 8) {
            int b = wave;
            float m = -3.4e38f;
            for (int j = lane; j < 999; j += 32) m = fmaxf(m, logits[b * NVP + j]);
#pragma unroll
            for (int d = 16; d >= 1; d >>= 1) m = fmaxf(m, __shfl_xor(m, d, 32));
            float sum = 0.f;
            for (int j = lane; j < 999; j += 32) sum += __expf(logits[b * NVP + j] - m);
#pragma unroll
            for (int d = 16; d >= 1; d >>= 1) sum += __shfl_xor(sum, d, 32);
            if (lane == 0) {
                float logZ = m + __logf(sum);
                int tgt = target[b * L_ + s + 1];
                atomicAdd(nll, logZ - logits[b * NVP + tgt]);
            }
        }
        __syncthreads();
    }

    if (tid == 0) out[0] = nll[0];
}

extern "C" void kernel_launch(void* const* d_in, const int* in_sizes, int n_in,
                              void* d_out, int out_size, void* d_ws, size_t ws_size,
                              hipStream_t stream) {
    const int*   target    = (const int*)  d_in[0];
    const float* enc_out   = (const float*)d_in[1];
    const float* enc_hid   = (const float*)d_in[2];
    const float* embedding = (const float*)d_in[3];
    const float* w_ih      = (const float*)d_in[4];
    const float* w_hh      = (const float*)d_in[5];
    const float* b_ih      = (const float*)d_in[6];
    const float* b_hh      = (const float*)d_in[7];
    const float* conv_w    = (const float*)d_in[8];
    const float* conv_b    = (const float*)d_in[9];
    const float* attn_w    = (const float*)d_in[10];
    const float* attn_b    = (const float*)d_in[11];
    const float* fc_w      = (const float*)d_in[12];
    const float* fc_b      = (const float*)d_in[13];
    (void)in_sizes; (void)n_in; (void)out_size; (void)ws_size;

    _Float16* w_ih16 = (_Float16*)d_ws;
    _Float16* w_hh16 = w_ih16 + WIH16_ELEMS;
    _Float16* fc16   = w_hh16 + WHH16_ELEMS;

    cvt_weights_kernel<<<(FC16_ELEMS + 255) / 256, 256, 0, stream>>>(
        w_ih, w_hh, fc_w, w_ih16, w_hh16, fc16);

    // allow >64KB dynamic LDS (gfx1250 WGP has 320KB); deterministic, not a stream op
    (void)hipFuncSetAttribute(reinterpret_cast<const void*>(decoder_persistent_kernel),
                              hipFuncAttributeMaxDynamicSharedMemorySize, SMEM_BYTES);

    decoder_persistent_kernel<<<1, 1024, SMEM_BYTES, stream>>>(
        target, enc_out, enc_hid, embedding, b_ih, b_hh,
        conv_w, conv_b, attn_w, attn_b, fc_b,
        w_ih16, w_hh16, fc16, (float*)d_out);
}
